// SwinTransformerBlock_9165460209912
// MI455X (gfx1250) — matrix operand order repaired and checked by reading, not compiled, not run on hardware
//
#include <hip/hip_runtime.h>
#include <hip/hip_bf16.h>

typedef __attribute__((ext_vector_type(16))) _Float16 v16h;
typedef __attribute__((ext_vector_type(8)))  _Float16 v8h;
typedef __attribute__((ext_vector_type(8)))  float    v8f;

#define BATCH 64
#define HDIM  56
#define WDIM  56
#define CDIM  256
#define NHEAD 8
#define WSZ   7
#define SSH   3
#define NTOK  49           // WSZ*WSZ
#define HIDD  1024
#define MROWS (BATCH*HDIM*WDIM)  // 200704

// ---------------------------------------------------------------------------
// CDNA5 async global->LDS copy (ASYNCcnt-tracked) + wait, via inline asm
// (portable across ROCm-7.2 / amdgpu-toolchain builtin arity differences).
// VDST = per-lane LDS byte offset, VADDR = per-lane 64-bit global address.
// ---------------------------------------------------------------------------
__device__ __forceinline__ void async_g2l_b128(const void* gptr, unsigned lds_off) {
    asm volatile("global_load_async_to_lds_b128 %0, %1, off"
                 :: "v"(lds_off), "v"((unsigned long long)(size_t)gptr)
                 : "memory");
}
__device__ __forceinline__ void wait_async0() {
    asm volatile("s_wait_asynccnt 0" ::: "memory");
}

// ---------------------------------------------------------------------------
// f32 -> f16 conversion for weights
// ---------------------------------------------------------------------------
__global__ void cvt_f16_kernel(const float* __restrict__ in, _Float16* __restrict__ out, int n) {
    int i = blockIdx.x * 256 + threadIdx.x;
    if (i < n) out[i] = (_Float16)in[i];
}

// ---------------------------------------------------------------------------
// LN1 + cyclic shift (-SSH) + window partition -> f16 rows in window layout
// row r = ((b*64 + wy*8 + wx)*49 + i*7 + j), reads x at h=(wy*7+i+SSH)%56 etc.
// one block per output row, 256 threads = 256 channels
// ---------------------------------------------------------------------------
__global__ void ln1_shift_window_kernel(const float* __restrict__ x,
                                        const float* __restrict__ w,
                                        const float* __restrict__ b,
                                        _Float16* __restrict__ Xw) {
    int r = blockIdx.x;
    int c = threadIdx.x;
    int bb  = r / (64 * NTOK);
    int rem = r % (64 * NTOK);
    int wl  = rem / NTOK;
    int t   = rem % NTOK;
    int wy = wl >> 3, wx = wl & 7;
    int i = t / WSZ, j = t % WSZ;
    int h  = (wy * WSZ + i + SSH) % HDIM;
    int ww = (wx * WSZ + j + SSH) % WDIM;
    const float* row = x + ((size_t)bb * (HDIM * WDIM) + h * WDIM + ww) * CDIM;

    float v = row[c];
    __shared__ float s1[256], s2[256];
    s1[c] = v; s2[c] = v * v;
    __syncthreads();
    for (int off = 128; off > 0; off >>= 1) {
        if (c < off) { s1[c] += s1[c + off]; s2[c] += s2[c + off]; }
        __syncthreads();
    }
    float mean = s1[0] * (1.0f / CDIM);
    float var  = s2[0] * (1.0f / CDIM) - mean * mean;
    float inv  = rsqrtf(var + 1e-5f);
    Xw[(size_t)r * CDIM + c] = (_Float16)((v - mean) * inv * w[c] + b[c]);
}

// ---------------------------------------------------------------------------
// Stage one 64x32 A-tile and 128x32 B-tile into LDS with async copies.
// 256 threads: A = 1 x b128/thread, B = 2 x b128/thread.
// ---------------------------------------------------------------------------
__device__ __forceinline__ void stage_tile(const _Float16* __restrict__ A,
                                           const _Float16* __restrict__ W,
                                           int K, int mBlock, int nBlock, int k0,
                                           _Float16* la, _Float16* lb, int tid) {
    int rowA = tid >> 2, chA = (tid & 3) * 8;
    async_g2l_b128(A + (size_t)(mBlock + rowA) * K + k0 + chA,
                   (unsigned)(size_t)(la + rowA * 32 + chA));
#pragma unroll
    for (int rr = 0; rr < 2; ++rr) {
        int r = tid + rr * 256;
        int rowB = r >> 2, chB = (r & 3) * 8;
        async_g2l_b128(W + (size_t)(nBlock + rowB) * K + k0 + chB,
                       (unsigned)(size_t)(lb + rowB * 32 + chB));
    }
}

// ---------------------------------------------------------------------------
// WMMA GEMM: out = A[M,K](f16) @ W[N,K](f16)^T + bias, templated epilogue.
// Block = 256 threads = 8 waves as 2x4; wave tile 32x32 (2x2 wmma frags);
// block tile 64(M) x 128(N); K staged 32-wide through double-buffered LDS
// via global_load_async_to_lds_b128 (ASYNCcnt). All shapes divide evenly.
// MODE 0: f16 out (+bias)          [QKV]
// MODE 1: f32 out (+bias)          [proj]
// MODE 2: f16 out, exact GELU      [fc1]
// MODE 3: f32 out += acc+bias      [fc2 -> d_out accumulate]
// ---------------------------------------------------------------------------
template <int MODE>
__global__ void gemm_wmma_kernel(const _Float16* __restrict__ A,
                                 const _Float16* __restrict__ W,
                                 const float* __restrict__ bias,
                                 void* __restrict__ out,
                                 int K, int Nout) {
    __shared__ __align__(32) _Float16 lA[2][64 * 32];
    __shared__ __align__(32) _Float16 lB[2][128 * 32];

    const int tid   = threadIdx.x;
    const int lane  = tid & 31;
    const int wid   = tid >> 5;
    const int waveM = wid >> 2;          // 0..1
    const int waveN = wid & 3;           // 0..3
    const int mBlock = blockIdx.y * 64;
    const int nBlock = blockIdx.x * 128;
    const int mWave  = waveM * 32;
    const int nWave  = waveN * 32;
    const int lm = lane & 15;
    const int g  = lane >> 4;

    v8f acc[2][2] = {};
    const int nk = K >> 5;

    stage_tile(A, W, K, mBlock, nBlock, 0, lA[0], lB[0], tid);
    wait_async0();
    __syncthreads();

    for (int kt = 0; kt < nk; ++kt) {
        const int cur = kt & 1;
        if (kt + 1 < nk)
            stage_tile(A, W, K, mBlock, nBlock, (kt + 1) << 5,
                       lA[cur ^ 1], lB[cur ^ 1], tid);

        v16h af[2], bf[2];
#pragma unroll
        for (int i = 0; i < 2; ++i) {
            const _Float16* pa = &lA[cur][(mWave + i * 16 + lm) * 32 + 16 * g];
            v8h lo = *(const v8h*)pa;
            v8h hi = *(const v8h*)(pa + 8);
#pragma unroll
            for (int t = 0; t < 8; ++t) { af[i][t] = lo[t]; af[i][t + 8] = hi[t]; }
        }
#pragma unroll
        for (int j = 0; j < 2; ++j) {
            const _Float16* pb = &lB[cur][(nWave + j * 16 + lm) * 32 + 16 * g];
            bf[j] = *(const v16h*)pb;
        }
#pragma unroll
        for (int i = 0; i < 2; ++i)
#pragma unroll
            for (int j = 0; j < 2; ++j)
                acc[i][j] = __builtin_amdgcn_wmma_f32_16x16x32_f16(
                    false, af[i], false, bf[j], (short)0, acc[i][j], false, false);

        wait_async0();      // next buffer landed; all waves done reading cur
        __syncthreads();
    }

#pragma unroll
    for (int i = 0; i < 2; ++i)
#pragma unroll
        for (int j = 0; j < 2; ++j) {
            int n = nBlock + nWave + j * 16 + lm;
            float bv = bias[n];
#pragma unroll
            for (int v = 0; v < 8; ++v) {
                int m = mBlock + mWave + i * 16 + g * 8 + v;
                float val = acc[i][j][v] + bv;
                size_t idx = (size_t)m * Nout + n;
                if (MODE == 0) {
                    ((_Float16*)out)[idx] = (_Float16)val;
                } else if (MODE == 1) {
                    ((float*)out)[idx] = val;
                } else if (MODE == 2) {
                    float gel = val * 0.5f * (1.0f + erff(val * 0.70710678118654752f));
                    ((_Float16*)out)[idx] = (_Float16)gel;
                } else {
                    ((float*)out)[idx] += val;
                }
            }
        }
}

// ---------------------------------------------------------------------------
// Windowed attention: one block per window (4096), loop over 8 heads.
// QKV rows: window layout, 768 cols = [q(8x32) | k(8x32) | v(8x32)].
// rel-pos bias from rpb_table[169,8]; shift mask computed analytically.
// ---------------------------------------------------------------------------
__global__ void attn_kernel(const _Float16* __restrict__ qkv,
                            const float* __restrict__ rpb,
                            _Float16* __restrict__ aout) {
    __shared__ float sq[NTOK * 32], sk[NTOK * 32], sv[NTOK * 32];
    __shared__ float sc[NTOK * NTOK];
    __shared__ int   cnt[NTOK];

    const int win = blockIdx.x;
    const int wl  = win & 63;
    const int wy  = wl >> 3, wx = wl & 7;
    const int tid = threadIdx.x;

    if (tid < NTOK) {
        int i = tid / WSZ, j = tid % WSZ;
        int gh = wy * WSZ + i, gw = wx * WSZ + j;
        int rh = (gh < HDIM - WSZ) ? 0 : ((gh < HDIM - SSH) ? 1 : 2);
        int rw = (gw < WDIM - WSZ) ? 0 : ((gw < WDIM - SSH) ? 1 : 2);
        cnt[tid] = rh * 3 + rw;
    }

    const _Float16* base = qkv + (size_t)win * NTOK * (3 * CDIM);

    for (int head = 0; head < NHEAD; ++head) {
        __syncthreads();
        // load q (scaled), k, v into LDS as f32
        for (int idx = tid; idx < NTOK * 32; idx += 256) {
            int n = idx >> 5, d = idx & 31;
            size_t ro = (size_t)n * (3 * CDIM) + head * 32 + d;
            sq[idx] = (float)base[ro]       * 0.17677669529663687f; // 1/sqrt(32)
            sk[idx] = (float)base[ro + 256];
            sv[idx] = (float)base[ro + 512];
        }
        __syncthreads();
        // scores + bias + mask
        for (int idx = tid; idx < NTOK * NTOK; idx += 256) {
            int n = idx / NTOK, m = idx % NTOK;
            float s = 0.0f;
#pragma unroll
            for (int d = 0; d < 32; ++d) s += sq[n * 32 + d] * sk[m * 32 + d];
            int i1 = n / WSZ, j1 = n % WSZ, i2 = m / WSZ, j2 = m % WSZ;
            int rpi = (i1 - i2 + WSZ - 1) * (2 * WSZ - 1) + (j1 - j2 + WSZ - 1);
            s += rpb[rpi * NHEAD + head];
            if (cnt[n] != cnt[m]) s -= 100.0f;
            sc[idx] = s;
        }
        __syncthreads();
        // softmax per row
        if (tid < NTOK) {
            float mx = -1e30f;
            for (int m = 0; m < NTOK; ++m) mx = fmaxf(mx, sc[tid * NTOK + m]);
            float sum = 0.0f;
            for (int m = 0; m < NTOK; ++m) {
                float e = __expf(sc[tid * NTOK + m] - mx);
                sc[tid * NTOK + m] = e;
                sum += e;
            }
            float inv = 1.0f / sum;
            for (int m = 0; m < NTOK; ++m) sc[tid * NTOK + m] *= inv;
        }
        __syncthreads();
        // out = P @ V
        for (int idx = tid; idx < NTOK * 32; idx += 256) {
            int n = idx >> 5, d = idx & 31;
            float o = 0.0f;
            for (int m = 0; m < NTOK; ++m) o += sc[n * NTOK + m] * sv[m * 32 + d];
            aout[((size_t)win * NTOK + n) * CDIM + head * 32 + d] = (_Float16)o;
        }
    }
}

// ---------------------------------------------------------------------------
// window reverse + unshift + residual (writes x2 to d_out) + LN2 -> f16 h2
// one block per token (b, h, w), 256 threads = channels
// ---------------------------------------------------------------------------
__global__ void residual_ln2_kernel(const float* __restrict__ x,
                                    const float* __restrict__ yp,   // f32, window layout
                                    const float* __restrict__ w2,
                                    const float* __restrict__ b2,
                                    float* __restrict__ xout,       // d_out
                                    _Float16* __restrict__ h2) {
    int tkn = blockIdx.x;
    int c   = threadIdx.x;
    int bb = tkn / (HDIM * WDIM);
    int hw = tkn % (HDIM * WDIM);
    int h = hw / WDIM, ww = hw % WDIM;
    int hs = (h  + HDIM - SSH) % HDIM;
    int ws = (ww + WDIM - SSH) % WDIM;
    int wy = hs / WSZ, i = hs % WSZ, wx = ws / WSZ, j = ws % WSZ;
    size_t r = ((size_t)bb * 64 + wy * 8 + wx) * NTOK + i * WSZ + j;

    float v = x[(size_t)tkn * CDIM + c] + yp[r * CDIM + c];
    xout[(size_t)tkn * CDIM + c] = v;

    __shared__ float s1[256], s2[256];
    s1[c] = v; s2[c] = v * v;
    __syncthreads();
    for (int off = 128; off > 0; off >>= 1) {
        if (c < off) { s1[c] += s1[c + off]; s2[c] += s2[c + off]; }
        __syncthreads();
    }
    float mean = s1[0] * (1.0f / CDIM);
    float var  = s2[0] * (1.0f / CDIM) - mean * mean;
    float inv  = rsqrtf(var + 1e-5f);
    h2[(size_t)tkn * CDIM + c] = (_Float16)((v - mean) * inv * w2[c] + b2[c]);
}

// ---------------------------------------------------------------------------
extern "C" void kernel_launch(void* const* d_in, const int* in_sizes, int n_in,
                              void* d_out, int out_size, void* d_ws, size_t ws_size,
                              hipStream_t stream) {
    (void)in_sizes; (void)n_in; (void)out_size; (void)ws_size;

    const float* x       = (const float*)d_in[0];
    const float* norm1_w = (const float*)d_in[1];
    const float* norm1_b = (const float*)d_in[2];
    const float* qkv_w   = (const float*)d_in[3];
    const float* qkv_b   = (const float*)d_in[4];
    const float* rpb     = (const float*)d_in[5];
    const float* proj_w  = (const float*)d_in[6];
    const float* proj_b  = (const float*)d_in[7];
    const float* norm2_w = (const float*)d_in[8];
    const float* norm2_b = (const float*)d_in[9];
    const float* fc1_w   = (const float*)d_in[10];
    const float* fc1_b   = (const float*)d_in[11];
    const float* fc2_w   = (const float*)d_in[12];
    const float* fc2_b   = (const float*)d_in[13];
    float* out = (float*)d_out;

    char* ws = (char*)d_ws;
    // --- f16 weights (front of workspace, < 2 MB) ---
    _Float16* w_qkv  = (_Float16*)ws;                       // 768*256
    _Float16* w_proj = w_qkv  + 768 * 256;                  // 256*256
    _Float16* w_fc1  = w_proj + 256 * 256;                  // 1024*256
    _Float16* w_fc2  = w_fc1  + 1024 * 256;                 // 256*1024
    // --- activation regions (lifetime-reused) ---
    const size_t REG = 117440512;                           // 112 MB
    _Float16* Xw   = (_Float16*)(ws + (size_t)(4 << 20));   // LN1 out / later h2
    _Float16* h2   = Xw;                                    // reuse (Xw dead after QKV gemm)
    _Float16* Aout = (_Float16*)(ws + (size_t)(4 << 20) + REG);
    char*     r1   = ws + (size_t)(4 << 20) + 2 * REG;      // 411 MB region
    _Float16* QKV  = (_Float16*)r1;                         // [M,768] f16
    float*    Yp   = (float*)r1;                            // [M,256] f32 (QKV dead)
    _Float16* Hm   = (_Float16*)r1;                         // [M,1024] f16 (Yp dead)

    // 0) weight conversion
    cvt_f16_kernel<<<(768 * 256 + 255) / 256, 256, 0, stream>>>(qkv_w, w_qkv, 768 * 256);
    cvt_f16_kernel<<<(256 * 256 + 255) / 256, 256, 0, stream>>>(proj_w, w_proj, 256 * 256);
    cvt_f16_kernel<<<(1024 * 256 + 255) / 256, 256, 0, stream>>>(fc1_w, w_fc1, 1024 * 256);
    cvt_f16_kernel<<<(256 * 1024 + 255) / 256, 256, 0, stream>>>(fc2_w, w_fc2, 256 * 1024);

    // 1) LN1 + shift + window partition
    ln1_shift_window_kernel<<<MROWS, 256, 0, stream>>>(x, norm1_w, norm1_b, Xw);

    // 2) QKV GEMM: [M,256] @ [768,256]^T -> f16 [M,768]
    gemm_wmma_kernel<0><<<dim3(768 / 128, MROWS / 64), 256, 0, stream>>>(
        Xw, w_qkv, qkv_b, QKV, 256, 768);

    // 3) windowed attention -> f16 [M,256] (window layout)
    attn_kernel<<<BATCH * 64, 256, 0, stream>>>(QKV, rpb, Aout);

    // 4) proj GEMM -> f32 [M,256] (window layout)
    gemm_wmma_kernel<1><<<dim3(256 / 128, MROWS / 64), 256, 0, stream>>>(
        Aout, w_proj, proj_b, Yp, 256, 256);

    // 5) reverse + residual (x2 -> d_out) + LN2 -> f16 h2
    residual_ln2_kernel<<<MROWS, 256, 0, stream>>>(x, Yp, norm2_w, norm2_b, out, h2);

    // 6) FC1 GEMM + GELU -> f16 [M,1024]
    gemm_wmma_kernel<2><<<dim3(1024 / 128, MROWS / 64), 256, 0, stream>>>(
        h2, w_fc1, fc1_b, Hm, 256, 1024);

    // 7) FC2 GEMM, accumulate into d_out
    gemm_wmma_kernel<3><<<dim3(256 / 128, MROWS / 64), 256, 0, stream>>>(
        Hm, w_fc2, fc2_b, out, 1024, 256);
}
